// DGCNN_28939489640608
// MI455X (gfx1250) — matrix operand order repaired and checked by reading, not compiled, not run on hardware
//
#include <hip/hip_runtime.h>
#include <hip/hip_bf16.h>

#define NPTS 4096
#define NB   8
#define KNN  20
#define EPSV 1e-5f

typedef _Float16 v16h __attribute__((ext_vector_type(16)));
typedef float    v8f  __attribute__((ext_vector_type(8)));

// Monotone float->uint encoding for atomicMax-based float max.
__device__ __forceinline__ unsigned fenc(float x){
  unsigned u = __float_as_uint(x);
  return (u & 0x80000000u) ? ~u : (u | 0x80000000u);
}
__device__ __forceinline__ float fdec(unsigned k){
  unsigned u = (k & 0x80000000u) ? (k & 0x7FFFFFFFu) : ~k;
  return __uint_as_float(u);
}

// WMMA fragment loaders (CDNA5 16-bit 16x16x32 layouts, wave32).
__device__ __forceinline__ v16h load_afrag(const _Float16* ab, int hh){
  v16h a;
  #pragma unroll
  for (int e = 0; e < 16; e++) a[e] = ab[e + 8 * hh + ((e >= 8) ? 8 : 0)];
  return a;
}
__device__ __forceinline__ v16h load_bfrag(const _Float16* wb, int hh){
  v16h b;
  #pragma unroll
  for (int e = 0; e < 16; e++) b[e] = wb[e + 8 * hh + ((e >= 8) ? 8 : 0)];
  return b;
}

// ---------------- transpose (B,6,N) -> (B,N,6) ----------------
__global__ void transpose_x_kernel(const float* __restrict__ x, float* __restrict__ x0){
  int t = blockIdx.x * blockDim.x + threadIdx.x;   // over B*N
  int b = t / NPTS, n = t % NPTS;
  #pragma unroll
  for (int c = 0; c < 6; c++)
    x0[(size_t)t * 6 + c] = x[((size_t)b * 6 + c) * NPTS + n];
}

// ---------------- f32 weights -> padded f16 [O][KP] ----------------
__global__ void convert_w_kernel(const float* __restrict__ W, _Float16* __restrict__ Wh,
                                 int O, int twoC, int KP){
  int t = blockIdx.x * blockDim.x + threadIdx.x;
  if (t >= O * KP) return;
  int o = t / KP, kk = t % KP;
  Wh[t] = (kk < twoC) ? (_Float16)W[o * twoC + kk] : (_Float16)0.f;
}

// ---------------- fused pairwise-distance + top-K ----------------
template<int C>
__global__ __launch_bounds__(256) void knn_kernel(const float* __restrict__ feat,
                                                  int* __restrict__ knn){
  extern __shared__ unsigned char dynsmem[];
  float* tile = (float*)dynsmem;        // [128][C]
  float* sqt  = tile + 128 * C;         // [128]
  const int TCOL = 128;
  const int tid = threadIdx.x;
  const int wgPerB = NPTS / 256;
  const int b = blockIdx.x / wgPerB;
  const int n = (blockIdx.x % wgPerB) * 256 + tid;
  const float* fb = feat + (size_t)b * NPTS * C;

  float rf[C];
  float rq = 0.f;
  #pragma unroll
  for (int c = 0; c < C; c++){ rf[c] = fb[(size_t)n * C + c]; rq += rf[c] * rf[c]; }

  float vals[KNN]; int idxs[KNN];
  #pragma unroll
  for (int i = 0; i < KNN; i++){ vals[i] = -3.4e38f; idxs[i] = 0; }
  float minval = -3.4e38f; int minpos = 0;

  for (int base = 0; base < NPTS; base += TCOL){
    __syncthreads();
    for (int i = tid; i < TCOL * C; i += 256) tile[i] = fb[(size_t)base * C + i];
    __syncthreads();
    if (tid < TCOL){
      float s = 0.f;
      #pragma unroll
      for (int c = 0; c < C; c++){ float v = tile[tid * C + c]; s += v * v; }
      sqt[tid] = s;
    }
    __syncthreads();
    for (int j = 0; j < TCOL; j++){
      float s = 0.f;
      #pragma unroll
      for (int c = 0; c < C; c++) s += rf[c] * tile[j * C + c];
      float d = 2.f * s - rq - sqt[j];     // negative squared distance (self == 0, max)
      if (d > minval){
        int m = base + j;
        #pragma unroll
        for (int i = 0; i < KNN; i++) if (i == minpos){ vals[i] = d; idxs[i] = m; }
        minval = vals[0]; minpos = 0;
        #pragma unroll
        for (int i = 1; i < KNN; i++) if (vals[i] < minval){ minval = vals[i]; minpos = i; }
      }
    }
  }
  int* kp = knn + ((size_t)b * NPTS + n) * KNN;
  #pragma unroll
  for (int i = 0; i < KNN; i++) kp[i] = idxs[i];
}

// ---------------- edge conv: gather -> WMMA GEMM -> BN/LReLU/max-k ----------------
template<int C, int O, int KP>
__global__ __launch_bounds__(256) void edge_conv_kernel(
    const float* __restrict__ feat, const int* __restrict__ knn,
    const _Float16* __restrict__ Wh,
    const float* __restrict__ gg, const float* __restrict__ bb,
    const float* __restrict__ rm, const float* __restrict__ rv,
    float* __restrict__ out){
  extern __shared__ unsigned char dynsmem[];
  float*    outLDS = (float*)dynsmem;               // [80][O]
  _Float16* eLDS   = (_Float16*)(outLDS + 80 * O);  // [80][KP]
  const int tid = threadIdx.x;
  const int wgPerB = NPTS / 4;
  const int b = blockIdx.x / wgPerB;
  const int nBase = (blockIdx.x % wgPerB) * 4;
  const float* fb = feat + (size_t)b * NPTS * C;
  const int* kb = knn + ((size_t)b * NPTS + nBase) * KNN;

  // build edge features [feat - ctr, ctr] (zero-padded to KP) as f16
  for (int t = tid; t < 80 * KP; t += 256){
    int row = t / KP, c = t % KP;
    int p = row / KNN, j = row % KNN;
    int n = nBase + p;
    int m = kb[p * KNN + j];
    float v;
    if (c < C)          v = fb[(size_t)m * C + c] - fb[(size_t)n * C + c];
    else if (c < 2 * C) v = fb[(size_t)n * C + (c - C)];
    else                v = 0.f;
    eLDS[t] = (_Float16)v;
  }
  __syncthreads();

  // WMMA GEMM: D[80][O] = edge[80][KP] * W^T[KP][O]  (B double-buffered)
  const int lane = tid & 31, wid = tid >> 5;
  const int r = lane & 15, hh = lane >> 4;
  const int NT = O / 16;
  for (int t = wid; t < 5 * NT; t += 8){
    int mt = t % 5, nt = t / 5;
    const _Float16* wrow = Wh + (size_t)(nt * 16 + r) * KP;
    v8f acc = {};
    v16h bcur = load_bfrag(wrow, hh);
    #pragma unroll
    for (int ko = 0; ko < KP; ko += 32){
      v16h a = load_afrag(eLDS + (mt * 16 + r) * KP + ko, hh);
      v16h bnxt = bcur;
      if (ko + 32 < KP) bnxt = load_bfrag(wrow + ko + 32, hh);
      acc = __builtin_amdgcn_wmma_f32_16x16x32_f16(false, a, false, bcur, (short)0, acc, false, false);
      bcur = bnxt;
    }
    #pragma unroll
    for (int e = 0; e < 8; e++)
      outLDS[(mt * 16 + e + 8 * hh) * O + nt * 16 + r] = acc[e];
  }
  __syncthreads();

  // BN -> LeakyReLU -> max over 20 neighbors
  for (int t = tid; t < 4 * O; t += 256){
    int p = t / O, o = t % O;
    float g = gg[o], be = bb[o], mu = rm[o], rs = rsqrtf(rv[o] + EPSV);
    float vmax = -3.4e38f;
    #pragma unroll
    for (int j = 0; j < KNN; j++){
      float v = outLDS[(p * KNN + j) * O + o];
      float h = g * (v - mu) * rs + be;
      h = (h > 0.f) ? h : 0.2f * h;
      vmax = fmaxf(vmax, h);
    }
    out[((size_t)b * NPTS + nBase + p) * O + o] = vmax;
  }
}

// ---------------- concat x1..x4 -> f16 (B,N,512) ----------------
__global__ void concat_kernel(const float* __restrict__ x1, const float* __restrict__ x2,
                              const float* __restrict__ x3, const float* __restrict__ x4,
                              _Float16* __restrict__ xc){
  size_t t = (size_t)blockIdx.x * blockDim.x + threadIdx.x;
  if (t >= (size_t)NB * NPTS * 512) return;
  int c = (int)(t & 511);
  size_t bn = t >> 9;
  float v;
  if      (c < 64)  v = x1[bn * 64  + c];
  else if (c < 128) v = x2[bn * 64  + (c - 64)];
  else if (c < 256) v = x3[bn * 128 + (c - 128)];
  else              v = x4[bn * 256 + (c - 256)];
  xc[t] = (_Float16)v;
}

__global__ void init_pool_kernel(unsigned* __restrict__ maxkey, float* __restrict__ sumbuf){
  int t = blockIdx.x * blockDim.x + threadIdx.x;
  if (t < NB * 1024){ maxkey[t] = fenc(-3.4e38f); sumbuf[t] = 0.f; }
}

// ---------------- h = lrelu(bn(xc @ W5^T)) fused with max/mean pooling ----------------
// xc staged to LDS via gfx1250 async copy path; pooling pre-reduced in LDS.
__global__ __launch_bounds__(256) void hgemm_pool_kernel(
    const _Float16* __restrict__ xc, const _Float16* __restrict__ W5h,
    const float* __restrict__ g5, const float* __restrict__ b5,
    const float* __restrict__ m5, const float* __restrict__ v5,
    unsigned* __restrict__ maxkey, float* __restrict__ sumbuf){
  extern __shared__ unsigned char dynsmem[];
  _Float16* sh   = (_Float16*)dynsmem;            // [64][512]  (64 KB)
  unsigned* lmax = (unsigned*)(sh + 64 * 512);    // [1024]
  float*    lsum = (float*)(lmax + 1024);         // [1024]
  const int tid = threadIdx.x;
  const size_t rowBase = (size_t)blockIdx.x * 64;

  // Async global->LDS staging of 64 rows x 512 f16 (16 B per op, 16 ops/thread).
  {
    const _Float16* gsrc = xc + rowBase * 512;
    for (int i = tid * 8; i < 64 * 512; i += 256 * 8){
      unsigned ldsOff = (unsigned)(size_t)(sh + i);          // flat LDS addr[31:0] == LDS offset
      const _Float16* gp = gsrc + i;
      asm volatile("global_load_async_to_lds_b128 %0, %1, off"
                   :: "v"(ldsOff), "v"(gp) : "memory");
    }
  }
  // Init LDS pooling accumulators while async copies are in flight.
  for (int i = tid; i < 1024; i += 256){ lmax[i] = fenc(-3.4e38f); lsum[i] = 0.f; }
  asm volatile("s_wait_asynccnt 0x0" ::: "memory");
  __syncthreads();

  const int lane = tid & 31, wid = tid >> 5;
  const int r = lane & 15, hh = lane >> 4;
  for (int t = wid; t < 4 * 64; t += 8){          // 4 M-tiles x 64 N-tiles
    int mt = t & 3, nt = t >> 2;
    const _Float16* wrow = W5h + (size_t)(nt * 16 + r) * 512;
    if (t + 8 < 4 * 64)
      __builtin_prefetch(W5h + (size_t)((((t + 8) >> 2) * 16) + r) * 512, 0, 2);
    v8f acc = {};
    v16h bcur = load_bfrag(wrow, hh);
    #pragma unroll
    for (int ko = 0; ko < 512; ko += 32){
      v16h a = load_afrag(sh + (mt * 16 + r) * 512 + ko, hh);
      v16h bnxt = bcur;
      if (ko + 32 < 512) bnxt = load_bfrag(wrow + ko + 32, hh);
      acc = __builtin_amdgcn_wmma_f32_16x16x32_f16(false, a, false, bcur, (short)0, acc, false, false);
      bcur = bnxt;
    }
    int o = nt * 16 + r;
    float g = g5[o], be = b5[o], mu = m5[o], rs = rsqrtf(v5[o] + EPSV);
    #pragma unroll
    for (int e = 0; e < 8; e++){
      float h = g * (acc[e] - mu) * rs + be;
      h = (h > 0.f) ? h : 0.2f * h;
      atomicMax(&lmax[o], fenc(h));   // LDS atomics (pre-reduction)
      atomicAdd(&lsum[o], h);
    }
  }
  __syncthreads();
  // One flush per WG: 64x fewer global atomics.
  const int bidx = (int)(rowBase >> 12);          // 4096 rows per batch
  for (int i = tid; i < 1024; i += 256){
    atomicMax(&maxkey[bidx * 1024 + i], lmax[i]);
    atomicAdd(&sumbuf[bidx * 1024 + i], lsum[i]);
  }
}

// ---------------- final MLP: (8,2048) -> 512 -> 256 -> 2 ----------------
__global__ __launch_bounds__(256) void final_mlp_kernel(
    const unsigned* __restrict__ maxkey, const float* __restrict__ sumbuf,
    const float* __restrict__ L1, const float* __restrict__ gl1, const float* __restrict__ bl1,
    const float* __restrict__ ml1, const float* __restrict__ vl1,
    const float* __restrict__ L2, const float* __restrict__ L2b,
    const float* __restrict__ gl2, const float* __restrict__ bl2,
    const float* __restrict__ ml2, const float* __restrict__ vl2,
    const float* __restrict__ L3, const float* __restrict__ L3b,
    float* __restrict__ outp){
  __shared__ float pooled[2048];
  __shared__ float o1[512];
  __shared__ float o2[256];
  const int b = blockIdx.x, tid = threadIdx.x;
  for (int o = tid; o < 1024; o += 256){
    pooled[o]        = fdec(maxkey[b * 1024 + o]);
    pooled[1024 + o] = sumbuf[b * 1024 + o] * (1.f / NPTS);
  }
  __syncthreads();
  for (int i = tid; i < 512; i += 256){
    float s = 0.f;
    for (int c = 0; c < 2048; c++) s += pooled[c] * L1[(size_t)i * 2048 + c];
    float h = gl1[i] * (s - ml1[i]) * rsqrtf(vl1[i] + EPSV) + bl1[i];
    o1[i] = (h > 0.f) ? h : 0.2f * h;
  }
  __syncthreads();
  for (int i = tid; i < 256; i += 256){
    float s = L2b[i];
    for (int c = 0; c < 512; c++) s += o1[c] * L2[(size_t)i * 512 + c];
    float h = gl2[i] * (s - ml2[i]) * rsqrtf(vl2[i] + EPSV) + bl2[i];
    o2[i] = (h > 0.f) ? h : 0.2f * h;
  }
  __syncthreads();
  if (tid < 2){
    float s = L3b[tid];
    for (int c = 0; c < 256; c++) s += o2[c] * L3[tid * 256 + c];
    outp[b * 2 + tid] = s;
  }
}

extern "C" void kernel_launch(void* const* d_in, const int* in_sizes, int n_in,
                              void* d_out, int out_size, void* d_ws, size_t ws_size,
                              hipStream_t stream){
  (void)in_sizes; (void)n_in; (void)out_size; (void)ws_size;
  const float* x  = (const float*)d_in[0];
  const float* W1 = (const float*)d_in[1];
  const float *g1 = (const float*)d_in[2],  *b1 = (const float*)d_in[3],
              *m1 = (const float*)d_in[4],  *v1 = (const float*)d_in[5];
  const float* W2 = (const float*)d_in[6];
  const float *g2 = (const float*)d_in[7],  *b2 = (const float*)d_in[8],
              *m2 = (const float*)d_in[9],  *v2 = (const float*)d_in[10];
  const float* W3 = (const float*)d_in[11];
  const float *g3 = (const float*)d_in[12], *b3 = (const float*)d_in[13],
              *m3 = (const float*)d_in[14], *v3 = (const float*)d_in[15];
  const float* W4 = (const float*)d_in[16];
  const float *g4 = (const float*)d_in[17], *b4 = (const float*)d_in[18],
              *m4 = (const float*)d_in[19], *v4 = (const float*)d_in[20];
  const float* W5 = (const float*)d_in[21];
  const float *g5 = (const float*)d_in[22], *b5 = (const float*)d_in[23],
              *m5 = (const float*)d_in[24], *v5 = (const float*)d_in[25];
  const float* L1 = (const float*)d_in[26];
  const float *gl1 = (const float*)d_in[27], *bl1 = (const float*)d_in[28],
              *ml1 = (const float*)d_in[29], *vl1 = (const float*)d_in[30];
  const float* L2  = (const float*)d_in[31];
  const float* L2b = (const float*)d_in[32];
  const float *gl2 = (const float*)d_in[33], *bl2 = (const float*)d_in[34],
              *ml2 = (const float*)d_in[35], *vl2 = (const float*)d_in[36];
  const float* L3  = (const float*)d_in[37];
  const float* L3b = (const float*)d_in[38];

  char* ws = (char*)d_ws;
  size_t off = 0;
  auto take = [&](size_t bytes) -> void* {
    void* p = ws + off;
    off = (off + bytes + 255) & ~(size_t)255;
    return p;
  };
  float* x0 = (float*)take((size_t)NB * NPTS * 6   * 4);
  float* x1 = (float*)take((size_t)NB * NPTS * 64  * 4);
  float* x2 = (float*)take((size_t)NB * NPTS * 64  * 4);
  float* x3 = (float*)take((size_t)NB * NPTS * 128 * 4);
  float* x4 = (float*)take((size_t)NB * NPTS * 256 * 4);
  int* knnb = (int*)take((size_t)NB * NPTS * KNN * 4);
  _Float16* xc16 = (_Float16*)take((size_t)NB * NPTS * 512 * 2);
  _Float16* W1h = (_Float16*)take((size_t)64   * 32  * 2);
  _Float16* W2h = (_Float16*)take((size_t)64   * 128 * 2);
  _Float16* W3h = (_Float16*)take((size_t)128  * 128 * 2);
  _Float16* W4h = (_Float16*)take((size_t)256  * 256 * 2);
  _Float16* W5h = (_Float16*)take((size_t)1024 * 512 * 2);
  unsigned* maxkey = (unsigned*)take((size_t)NB * 1024 * 4);
  float*    sumbuf = (float*)take((size_t)NB * 1024 * 4);

  transpose_x_kernel<<<NB * NPTS / 256, 256, 0, stream>>>(x, x0);
  convert_w_kernel<<<(64 * 32 + 255) / 256,    256, 0, stream>>>(W1, W1h, 64,   12,  32);
  convert_w_kernel<<<(64 * 128 + 255) / 256,   256, 0, stream>>>(W2, W2h, 64,   128, 128);
  convert_w_kernel<<<(128 * 128 + 255) / 256,  256, 0, stream>>>(W3, W3h, 128,  128, 128);
  convert_w_kernel<<<(256 * 256 + 255) / 256,  256, 0, stream>>>(W4, W4h, 256,  256, 256);
  convert_w_kernel<<<(1024 * 512 + 255) / 256, 256, 0, stream>>>(W5, W5h, 1024, 512, 512);

  const int knnGrid  = NB * (NPTS / 256);
  const int edgeGrid = NB * (NPTS / 4);

  knn_kernel<6><<<knnGrid, 256, 128 * 6 * 4 + 128 * 4, stream>>>(x0, knnb);
  edge_conv_kernel<6, 64, 32><<<edgeGrid, 256, 80 * 64 * 4 + 80 * 32 * 2, stream>>>(
      x0, knnb, W1h, g1, b1, m1, v1, x1);

  knn_kernel<64><<<knnGrid, 256, 128 * 64 * 4 + 128 * 4, stream>>>(x1, knnb);
  edge_conv_kernel<64, 64, 128><<<edgeGrid, 256, 80 * 64 * 4 + 80 * 128 * 2, stream>>>(
      x1, knnb, W2h, g2, b2, m2, v2, x2);

  knn_kernel<64><<<knnGrid, 256, 128 * 64 * 4 + 128 * 4, stream>>>(x2, knnb);
  edge_conv_kernel<64, 128, 128><<<edgeGrid, 256, 80 * 128 * 4 + 80 * 128 * 2, stream>>>(
      x2, knnb, W3h, g3, b3, m3, v3, x3);

  knn_kernel<128><<<knnGrid, 256, 128 * 128 * 4 + 128 * 4, stream>>>(x3, knnb);
  edge_conv_kernel<128, 256, 256><<<edgeGrid, 256, 80 * 256 * 4 + 80 * 256 * 2, stream>>>(
      x3, knnb, W4h, g4, b4, m4, v4, x4);

  concat_kernel<<<(int)(((size_t)NB * NPTS * 512 + 255) / 256), 256, 0, stream>>>(x1, x2, x3, x4, xc16);
  init_pool_kernel<<<(NB * 1024 + 255) / 256, 256, 0, stream>>>(maxkey, sumbuf);
  hgemm_pool_kernel<<<NB * NPTS / 64, 256, 64 * 512 * 2 + 1024 * 4 + 1024 * 4, stream>>>(
      xc16, W5h, g5, b5, m5, v5, maxkey, sumbuf);
  final_mlp_kernel<<<NB, 256, 0, stream>>>(maxkey, sumbuf,
      L1, gl1, bl1, ml1, vl1, L2, L2b, gl2, bl2, ml2, vl2, L3, L3b, (float*)d_out);
}
